// AdaptiveGNN_82703890251959
// MI455X (gfx1250) — compile-verified, matrix operand order checked by reading, not run on hardware
//
#include <hip/hip_runtime.h>
#include <hip/hip_bf16.h>
#include <math.h>

#define NN 50000
#define NE 400000
#define DF 64
// extended edge list = edges + self loops
#define NEX (NE + NN)

typedef __attribute__((ext_vector_type(16))) _Float16 v16h;
typedef __attribute__((ext_vector_type(8)))  float    v8f;

// ---------------- helpers ----------------
__device__ __forceinline__ unsigned fkey(float f) {
    unsigned u = __float_as_uint(f);
    return (u & 0x80000000u) ? ~u : (u | 0x80000000u);
}
__device__ __forceinline__ float funkey(unsigned k) {
    unsigned u = (k & 0x80000000u) ? (k & 0x7FFFFFFFu) : ~k;
    return __uint_as_float(u);
}
__device__ __forceinline__ float lrelu(float v) { return v > 0.f ? v : 0.2f * v; }

__device__ __forceinline__ void edge_rc(const int* __restrict__ ei, long long e,
                                        int& r, int& c) {
    if (e < NE) { r = ei[e]; c = ei[NE + e]; }
    else        { r = (int)(e - NE); c = r; }
}

// ---------------- fills ----------------
__global__ void k_fill_f32(float* p, float v, long long n) {
    long long i = (long long)blockIdx.x * blockDim.x + threadIdx.x;
    if (i < n) p[i] = v;
}
__global__ void k_fill_u32(unsigned* p, unsigned v, long long n) {
    long long i = (long long)blockIdx.x * blockDim.x + threadIdx.x;
    if (i < n) p[i] = v;
}

// ---------------- WMMA GEMM: C[NN,NC] = A[NN,K] @ W  (W given transposed: Wt[NC,K]) ----
// Compile-time K/NC/bias/accum: straight-line body, fully unrolled K loop.
// NN = 50000 = 3125*16 exactly, so no M-edge handling is needed.
template <int K, int NC, bool HAS_BIAS, bool ACCUM>
__global__ void k_gemm_wmma(const float* __restrict__ A,
                            const float* __restrict__ Wt,
                            const float* __restrict__ bias,
                            float* __restrict__ C) {
    const int lane = threadIdx.x & 31;
    const int wave = threadIdx.x >> 5;
    constexpr int nT = NC / 16;
    constexpr int mT = NN / 16;
    const int tile = blockIdx.x * 8 + wave;
    if (tile >= mT * nT) return;
    const int mt = tile / nT;
    const int nt = tile % nT;

    const int half_sel = lane >> 4;        // 0: K 0..7 / 16..23 ; 1: K 8..15 / 24..31
    const int lrow = lane & 15;
    const int kb = half_sel * 8;

    const float* arow = A  + (size_t)(mt * 16 + lrow) * K + kb;
    const float* brow = Wt + (size_t)(nt * 16 + lrow) * K + kb;

    v8f acc = {0.f, 0.f, 0.f, 0.f, 0.f, 0.f, 0.f, 0.f};
#pragma unroll
    for (int k0 = 0; k0 < K; k0 += 32) {
        // 32-byte-aligned contiguous chunks -> global_load_b128 pairs
        const float4 a0 = *(const float4*)(arow + k0);
        const float4 a1 = *(const float4*)(arow + k0 + 4);
        const float4 a2 = *(const float4*)(arow + k0 + 16);
        const float4 a3 = *(const float4*)(arow + k0 + 20);
        const float4 b0 = *(const float4*)(brow + k0);
        const float4 b1 = *(const float4*)(brow + k0 + 4);
        const float4 b2 = *(const float4*)(brow + k0 + 16);
        const float4 b3 = *(const float4*)(brow + k0 + 20);
        v16h a, b;
        a[0]  = (_Float16)a0.x; a[1]  = (_Float16)a0.y; a[2]  = (_Float16)a0.z; a[3]  = (_Float16)a0.w;
        a[4]  = (_Float16)a1.x; a[5]  = (_Float16)a1.y; a[6]  = (_Float16)a1.z; a[7]  = (_Float16)a1.w;
        a[8]  = (_Float16)a2.x; a[9]  = (_Float16)a2.y; a[10] = (_Float16)a2.z; a[11] = (_Float16)a2.w;
        a[12] = (_Float16)a3.x; a[13] = (_Float16)a3.y; a[14] = (_Float16)a3.z; a[15] = (_Float16)a3.w;
        b[0]  = (_Float16)b0.x; b[1]  = (_Float16)b0.y; b[2]  = (_Float16)b0.z; b[3]  = (_Float16)b0.w;
        b[4]  = (_Float16)b1.x; b[5]  = (_Float16)b1.y; b[6]  = (_Float16)b1.z; b[7]  = (_Float16)b1.w;
        b[8]  = (_Float16)b2.x; b[9]  = (_Float16)b2.y; b[10] = (_Float16)b2.z; b[11] = (_Float16)b2.w;
        b[12] = (_Float16)b3.x; b[13] = (_Float16)b3.y; b[14] = (_Float16)b3.z; b[15] = (_Float16)b3.w;
        acc = __builtin_amdgcn_wmma_f32_16x16x32_f16(false, a, false, b,
                                                     (short)0, acc, false, false);
    }

    const int col = nt * 16 + lrow;
    const float bv = HAS_BIAS ? bias[col] : 0.f;
    // C/D layout: VGPR v holds row (v + 8*half_sel) of the tile, col = lane&15
    float* crow = C + (size_t)(mt * 16 + half_sel * 8) * NC + col;
#pragma unroll
    for (int v = 0; v < 8; ++v) {
        float val = acc[v] + bv;
        if (ACCUM) crow[(size_t)v * NC] += val;
        else       crow[(size_t)v * NC]  = val;
    }
}

// ---------------- small prep ----------------
__global__ void k_transpose(const float* __restrict__ W, float* __restrict__ Wt,
                            int rows, int cols) {
    int i = blockIdx.x * blockDim.x + threadIdx.x;
    if (i >= rows * cols) return;
    int r = i / cols, c = i % cols;
    Wt[(size_t)c * rows + r] = W[i];
}

__global__ void k_colmean(const float* __restrict__ x, float* __restrict__ g) {
    __shared__ float red[256];
    int col = blockIdx.x, t = threadIdx.x;
    float s = 0.f;
    for (int i = t; i < NN; i += 256) s += x[(size_t)i * DF + col];
    red[t] = s; __syncthreads();
    for (int off = 128; off > 0; off >>= 1) {
        if (t < off) red[t] += red[t + off];
        __syncthreads();
    }
    if (t == 0) g[col] = red[0] / (float)NN;
}

__global__ void k_gate(const float* __restrict__ g,
                       const float* __restrict__ w1, const float* __restrict__ b1,
                       const float* __restrict__ w2, const float* __restrict__ b2,
                       float* __restrict__ wout) {
    __shared__ float h[64];
    __shared__ float z[3];
    int t = threadIdx.x;  // 64 threads
    float acc = b1[t];
    for (int i = 0; i < 64; ++i) acc += g[i] * w1[i * 64 + t];
    h[t] = fmaxf(acc, 0.f);
    __syncthreads();
    if (t < 3) {
        float zz = b2[t];
        for (int j = 0; j < 64; ++j) zz += h[j] * w2[j * 3 + t];
        z[t] = zz;
    }
    __syncthreads();
    if (t == 0) {
        float m = fmaxf(z[0], fmaxf(z[1], z[2]));
        float e0 = expf(z[0] - m), e1 = expf(z[1] - m), e2 = expf(z[2] - m);
        float s = e0 + e1 + e2;
        wout[0] = e0 / s; wout[1] = e1 / s; wout[2] = e2 / s;
    }
}

// ---------------- degree / norm ----------------
__global__ void k_degree(const int* __restrict__ ei, float* __restrict__ deg) {
    long long e = (long long)blockIdx.x * blockDim.x + threadIdx.x;
    if (e >= NEX) return;
    int r, c; edge_rc(ei, e, r, c);
    atomicAdd(deg + c, 1.0f);
}
__global__ void k_dis(const float* __restrict__ deg, float* __restrict__ dis) {
    int i = blockIdx.x * blockDim.x + threadIdx.x;
    if (i < NN) dis[i] = rsqrtf(deg[i]);   // deg >= 1 (self loop)
}

// ---------------- GCN aggregate ----------------
__global__ void k_gcn_agg(const int* __restrict__ ei, const float* __restrict__ h,
                          const float* __restrict__ dis, float* __restrict__ out) {
    long long gid = (long long)blockIdx.x * blockDim.x + threadIdx.x;
    if (gid >= (long long)NEX * DF) return;
    long long e = gid >> 6;
    int f = (int)(gid & 63);
    int r, c; edge_rc(ei, e, r, c);
    float v = h[(size_t)r * DF + f] * dis[r] * dis[c];
    atomicAdd(out + (size_t)c * DF + f, v);
}

__global__ void k_bnrelu(float* __restrict__ h, const float* __restrict__ b,
                         const float* __restrict__ gamma, const float* __restrict__ beta) {
    long long gid = (long long)blockIdx.x * blockDim.x + threadIdx.x;
    if (gid >= (long long)NN * DF) return;
    int f = (int)(gid & 63);
    float v = (h[gid] + b[f]) * (gamma[f] * rsqrtf(1.0f + 1e-5f)) + beta[f];
    h[gid] = fmaxf(v, 0.f);
}

// ---------------- GAT ----------------
__global__ void k_gat_scores(const float* __restrict__ h,
                             const float* __restrict__ att_src,
                             const float* __restrict__ att_dst,
                             float* __restrict__ as_, float* __restrict__ ad_, int H) {
    int gid = blockIdx.x * blockDim.x + threadIdx.x;
    if (gid >= NN * H) return;
    int n = gid / H, hh = gid % H;
    const float* hp = h + ((size_t)n * H + hh) * DF;
    float s = 0.f, d = 0.f;
    for (int f = 0; f < DF; ++f) {
        s += hp[f] * att_src[hh * DF + f];
        d += hp[f] * att_dst[hh * DF + f];
    }
    as_[gid] = s; ad_[gid] = d;
}

__global__ void k_gat_max(const int* __restrict__ ei, const float* __restrict__ as_,
                          const float* __restrict__ ad_, unsigned* __restrict__ mkey, int H) {
    long long gid = (long long)blockIdx.x * blockDim.x + threadIdx.x;
    if (gid >= (long long)NEX * H) return;
    long long e = gid / H;
    int hh = (int)(gid % H);
    int r, c; edge_rc(ei, e, r, c);
    float ev = lrelu(as_[r * H + hh] + ad_[c * H + hh]);
    atomicMax(mkey + c * H + hh, fkey(ev));
}

__global__ void k_gat_expsum(const int* __restrict__ ei, const float* __restrict__ as_,
                             const float* __restrict__ ad_, const unsigned* __restrict__ mkey,
                             float* __restrict__ denom, float* __restrict__ ew, int H) {
    long long gid = (long long)blockIdx.x * blockDim.x + threadIdx.x;
    if (gid >= (long long)NEX * H) return;
    long long e = gid / H;
    int hh = (int)(gid % H);
    int r, c; edge_rc(ei, e, r, c);
    float ev = lrelu(as_[r * H + hh] + ad_[c * H + hh]);
    float w = expf(ev - funkey(mkey[c * H + hh]));
    ew[gid] = w;
    atomicAdd(denom + c * H + hh, w);
}

__global__ void k_gat_acc(const int* __restrict__ ei, const float* __restrict__ hsrc,
                          const float* __restrict__ ew, float* __restrict__ S, int H) {
    long long gid = (long long)blockIdx.x * blockDim.x + threadIdx.x;
    if (gid >= (long long)NEX * H * DF) return;
    int f = (int)(gid & 63);
    long long t = gid >> 6;
    long long e = t / H;
    int hh = (int)(t % H);
    int r, c; edge_rc(ei, e, r, c);
    float v = hsrc[((size_t)r * H + hh) * DF + f] * ew[e * H + hh];
    atomicAdd(S + ((size_t)c * H + hh) * DF + f, v);
}

__global__ void k_gat_norm(const float* __restrict__ S, const float* __restrict__ denom,
                           const float* __restrict__ bias, float* __restrict__ out,
                           int H, int do_elu) {
    long long gid = (long long)blockIdx.x * blockDim.x + threadIdx.x;
    if (gid >= (long long)NN * H * DF) return;
    int f = (int)(gid & 63);
    long long t = gid >> 6;
    int n = (int)(t / H), hh = (int)(t % H);
    float v = S[gid] / denom[n * H + hh] + bias[hh * DF + f];
    if (do_elu) v = v > 0.f ? v : expm1f(v);
    out[gid] = v;
}

// ---------------- SAGE ----------------
__global__ void k_sage_scatter(const int* __restrict__ ei, const float* __restrict__ x,
                               float* __restrict__ s) {
    long long gid = (long long)blockIdx.x * blockDim.x + threadIdx.x;
    if (gid >= (long long)NE * DF) return;
    long long e = gid >> 6;
    int f = (int)(gid & 63);
    int r = ei[e], c = ei[NE + e];
    atomicAdd(s + (size_t)c * DF + f, x[(size_t)r * DF + f]);
}
__global__ void k_sage_mean(float* __restrict__ s, const float* __restrict__ deg) {
    long long gid = (long long)blockIdx.x * blockDim.x + threadIdx.x;
    if (gid >= (long long)NN * DF) return;
    int n = (int)(gid >> 6);
    s[gid] /= fmaxf(deg[n] - 1.0f, 1.0f);   // deg includes self loop
}
__global__ void k_l2norm(float* __restrict__ h, int do_relu) {
    int n = blockIdx.x * blockDim.x + threadIdx.x;
    if (n >= NN) return;
    float* row = h + (size_t)n * DF;
    float s = 0.f;
    for (int f = 0; f < DF; ++f) s += row[f] * row[f];
    float inv = 1.0f / fmaxf(sqrtf(s), 1e-12f);
    for (int f = 0; f < DF; ++f) {
        float v = row[f] * inv;
        row[f] = do_relu ? fmaxf(v, 0.f) : v;
    }
}

// ---------------- final combine ----------------
__global__ void k_combine(float* __restrict__ dst, const float* __restrict__ src,
                          const float* __restrict__ bias, const float* __restrict__ wg,
                          int widx, int acc) {
    long long gid = (long long)blockIdx.x * blockDim.x + threadIdx.x;
    if (gid >= (long long)NN * DF) return;
    int f = (int)(gid & 63);
    float v = src[gid] + (bias ? bias[f] : 0.f);
    v *= wg[widx];
    dst[gid] = acc ? (dst[gid] + v) : v;
}

// ---------------- launcher ----------------
static inline unsigned gb(long long n, int b = 256) { return (unsigned)((n + b - 1) / b); }
static inline unsigned gemm_blocks(int NC) { return (unsigned)(((NN / 16) * (NC / 16) + 7) / 8); }

extern "C" void kernel_launch(void* const* d_in, const int* in_sizes, int n_in,
                              void* d_out, int out_size, void* d_ws, size_t ws_size,
                              hipStream_t stream) {
    (void)in_sizes; (void)n_in; (void)out_size; (void)ws_size;
    const float* x        = (const float*)d_in[0];
    const int*   ei       = (const int*)  d_in[1];
    const float* gate_w1  = (const float*)d_in[2];
    const float* gate_b1  = (const float*)d_in[3];
    const float* gate_w2  = (const float*)d_in[4];
    const float* gate_b2  = (const float*)d_in[5];
    const float* gcn_w1   = (const float*)d_in[6];
    const float* gcn_b1   = (const float*)d_in[7];
    const float* bn_gamma = (const float*)d_in[8];
    const float* bn_beta  = (const float*)d_in[9];
    const float* gcn_w2   = (const float*)d_in[10];
    const float* gcn_b2   = (const float*)d_in[11];
    const float* gat_w1   = (const float*)d_in[12];
    const float* att_s1   = (const float*)d_in[13];
    const float* att_d1   = (const float*)d_in[14];
    const float* gat_b1   = (const float*)d_in[15];
    const float* gat_w2   = (const float*)d_in[16];
    const float* att_s2   = (const float*)d_in[17];
    const float* att_d2   = (const float*)d_in[18];
    const float* gat_b2   = (const float*)d_in[19];
    const float* sage_wl1 = (const float*)d_in[20];
    const float* sage_bl1 = (const float*)d_in[21];
    const float* sage_wr1 = (const float*)d_in[22];
    const float* sage_wl2 = (const float*)d_in[23];
    const float* sage_bl2 = (const float*)d_in[24];
    const float* sage_wr2 = (const float*)d_in[25];
    float* out = (float*)d_out;

    // workspace layout (floats)
    float* p  = (float*)d_ws;
    float* B0 = p; p += (size_t)NN * 256;   // GAT h1 / generic
    float* B1 = p; p += (size_t)NN * 256;   // GAT S1
    float* B2 = p; p += (size_t)NN * DF;    // generic
    float* B3 = p; p += (size_t)NN * DF;    // generic
    float* ew = p; p += (size_t)NEX * 4;
    float* deg = p; p += NN;
    float* dis = p; p += NN;
    float* as_ = p; p += NN * 4;
    float* ad_ = p; p += NN * 4;
    unsigned* mkey = (unsigned*)p; p += NN * 4;
    float* denom = p; p += NN * 4;
    float* wt_gcn1 = p; p += 64 * 64;
    float* wt_gcn2 = p; p += 64 * 64;
    float* wt_gat1 = p; p += 64 * 256;      // [256][64]
    float* wt_gat2 = p; p += 64 * 256;      // [64][256]
    float* wt_sl1  = p; p += 64 * 64;
    float* wt_sr1  = p; p += 64 * 64;
    float* wt_sl2  = p; p += 64 * 64;
    float* wt_sr2  = p; p += 64 * 64;
    float* gvec  = p; p += 64;
    float* wgate = p; p += 4;

    const unsigned KEYNEG = 0x00800000u;    // fkey(-FLT_MAX)

    // weight transposes (Wt[n][k])
    k_transpose<<<gb(64 * 64), 256, 0, stream>>>(gcn_w1, wt_gcn1, 64, 64);
    k_transpose<<<gb(64 * 64), 256, 0, stream>>>(gcn_w2, wt_gcn2, 64, 64);
    k_transpose<<<gb(64 * 256), 256, 0, stream>>>(gat_w1, wt_gat1, 64, 256);
    k_transpose<<<gb(256 * 64), 256, 0, stream>>>(gat_w2, wt_gat2, 256, 64);
    k_transpose<<<gb(64 * 64), 256, 0, stream>>>(sage_wl1, wt_sl1, 64, 64);
    k_transpose<<<gb(64 * 64), 256, 0, stream>>>(sage_wr1, wt_sr1, 64, 64);
    k_transpose<<<gb(64 * 64), 256, 0, stream>>>(sage_wl2, wt_sl2, 64, 64);
    k_transpose<<<gb(64 * 64), 256, 0, stream>>>(sage_wr2, wt_sr2, 64, 64);

    // gate
    k_colmean<<<64, 256, 0, stream>>>(x, gvec);
    k_gate<<<1, 64, 0, stream>>>(gvec, gate_w1, gate_b1, gate_w2, gate_b2, wgate);

    // degree (with self loops) + symmetric norm
    k_fill_f32<<<gb(NN), 256, 0, stream>>>(deg, 0.f, NN);
    k_degree<<<gb(NEX), 256, 0, stream>>>(ei, deg);
    k_dis<<<gb(NN), 256, 0, stream>>>(deg, dis);

    // ---------- GCN branch ----------
    k_gemm_wmma<64, 64, false, false><<<gemm_blocks(64), 256, 0, stream>>>(x, wt_gcn1, nullptr, B2);
    k_fill_f32<<<gb((long long)NN * DF), 256, 0, stream>>>(B3, 0.f, (long long)NN * DF);
    k_gcn_agg<<<gb((long long)NEX * DF), 256, 0, stream>>>(ei, B2, dis, B3);
    k_bnrelu<<<gb((long long)NN * DF), 256, 0, stream>>>(B3, gcn_b1, bn_gamma, bn_beta);
    k_gemm_wmma<64, 64, false, false><<<gemm_blocks(64), 256, 0, stream>>>(B3, wt_gcn2, nullptr, B2);
    k_fill_f32<<<gb((long long)NN * DF), 256, 0, stream>>>(B3, 0.f, (long long)NN * DF);
    k_gcn_agg<<<gb((long long)NEX * DF), 256, 0, stream>>>(ei, B2, dis, B3);
    k_combine<<<gb((long long)NN * DF), 256, 0, stream>>>(out, B3, gcn_b2, wgate, 0, 0);

    // ---------- GAT branch, layer 1 (H=4, concat+ELU) ----------
    k_gemm_wmma<64, 256, false, false><<<gemm_blocks(256), 256, 0, stream>>>(x, wt_gat1, nullptr, B0);
    k_gat_scores<<<gb(NN * 4), 256, 0, stream>>>(B0, att_s1, att_d1, as_, ad_, 4);
    k_fill_u32<<<gb(NN * 4), 256, 0, stream>>>(mkey, KEYNEG, NN * 4);
    k_gat_max<<<gb((long long)NEX * 4), 256, 0, stream>>>(ei, as_, ad_, mkey, 4);
    k_fill_f32<<<gb(NN * 4), 256, 0, stream>>>(denom, 0.f, NN * 4);
    k_gat_expsum<<<gb((long long)NEX * 4), 256, 0, stream>>>(ei, as_, ad_, mkey, denom, ew, 4);
    k_fill_f32<<<gb((long long)NN * 256), 256, 0, stream>>>(B1, 0.f, (long long)NN * 256);
    k_gat_acc<<<gb((long long)NEX * 4 * DF), 256, 0, stream>>>(ei, B0, ew, B1, 4);
    k_gat_norm<<<gb((long long)NN * 256), 256, 0, stream>>>(B1, denom, gat_b1, B0, 4, 1);

    // ---------- GAT layer 2 (H=1, mean) ----------
    k_gemm_wmma<256, 64, false, false><<<gemm_blocks(64), 256, 0, stream>>>(B0, wt_gat2, nullptr, B2);
    k_gat_scores<<<gb(NN), 256, 0, stream>>>(B2, att_s2, att_d2, as_, ad_, 1);
    k_fill_u32<<<gb(NN), 256, 0, stream>>>(mkey, KEYNEG, NN);
    k_gat_max<<<gb(NEX), 256, 0, stream>>>(ei, as_, ad_, mkey, 1);
    k_fill_f32<<<gb(NN), 256, 0, stream>>>(denom, 0.f, NN);
    k_gat_expsum<<<gb(NEX), 256, 0, stream>>>(ei, as_, ad_, mkey, denom, ew, 1);
    k_fill_f32<<<gb((long long)NN * DF), 256, 0, stream>>>(B3, 0.f, (long long)NN * DF);
    k_gat_acc<<<gb((long long)NEX * DF), 256, 0, stream>>>(ei, B2, ew, B3, 1);
    k_gat_norm<<<gb((long long)NN * DF), 256, 0, stream>>>(B3, denom, gat_b2, B3, 1, 0);
    k_combine<<<gb((long long)NN * DF), 256, 0, stream>>>(out, B3, nullptr, wgate, 1, 1);

    // ---------- SAGE branch, layer 1 ----------
    k_fill_f32<<<gb((long long)NN * DF), 256, 0, stream>>>(B3, 0.f, (long long)NN * DF);
    k_sage_scatter<<<gb((long long)NE * DF), 256, 0, stream>>>(ei, x, B3);
    k_sage_mean<<<gb((long long)NN * DF), 256, 0, stream>>>(B3, deg);
    k_gemm_wmma<64, 64, true, false><<<gemm_blocks(64), 256, 0, stream>>>(B3, wt_sl1, sage_bl1, B2);
    k_gemm_wmma<64, 64, false, true><<<gemm_blocks(64), 256, 0, stream>>>(x, wt_sr1, nullptr, B2);
    k_l2norm<<<gb(NN), 256, 0, stream>>>(B2, 1);

    // ---------- SAGE layer 2 ----------
    k_fill_f32<<<gb((long long)NN * DF), 256, 0, stream>>>(B3, 0.f, (long long)NN * DF);
    k_sage_scatter<<<gb((long long)NE * DF), 256, 0, stream>>>(ei, B2, B3);
    k_sage_mean<<<gb((long long)NN * DF), 256, 0, stream>>>(B3, deg);
    k_gemm_wmma<64, 64, true, false><<<gemm_blocks(64), 256, 0, stream>>>(B3, wt_sl2, sage_bl2, B0);
    k_gemm_wmma<64, 64, false, true><<<gemm_blocks(64), 256, 0, stream>>>(B2, wt_sr2, nullptr, B0);
    k_l2norm<<<gb(NN), 256, 0, stream>>>(B0, 0);
    k_combine<<<gb((long long)NN * DF), 256, 0, stream>>>(out, B0, nullptr, wgate, 2, 1);
}
